// ASIC_9354438771385
// MI455X (gfx1250) — matrix operand order repaired and checked by reading, not compiled, not run on hardware
//
#include <hip/hip_runtime.h>

typedef __attribute__((ext_vector_type(16))) _Float16 v16h;
typedef __attribute__((ext_vector_type(8)))  float    v8f;

#define NPIX 4096   // 64*64
#define NB   256    // batch / column length
#define NPAT 16

__device__ __forceinline__ float clamp01(float v) {
    return fminf(fmaxf(v, 0.0f), 1.0f);
}

// One block per (h,w) pixel column. 256 threads = 8 waves (wave32).
// Each wave owns 32 consecutive b-rows; einsum over 16 patterns done with
// two V_WMMA_F32_16X16X32_F16 per wave per layer (K padded 16->32).
__global__ __launch_bounds__(256) void lgn_forward_kernel(
    const float* __restrict__ x,        // (256, 4096)
    const float* __restrict__ gates,    // (4, 16, 4096)
    float* __restrict__ out,            // (256, 4096)
    float* __restrict__ blocksum)       // (4096,) partial reg sums
{
    const int pix       = blockIdx.x;
    const int tid       = threadIdx.x;
    const int lane      = tid & 31;
    const int waveBase  = tid & ~31;     // first b-row of this wave's 32 rows
    const int halfSel   = lane >> 4;     // 0 -> patterns 0..7, 1 -> patterns 8..15
    const int rowInTile = lane & 15;     // A-matrix row M within 16-row tile

    __shared__ float    buf[2][NB];      // double-buffered column values
    __shared__ _Float16 tvals[NPAT];     // sigmoid(gates) for this layer/pixel
    __shared__ float    red[NB];

    // Load column: cur[b] = x[b, pix]
    buf[0][tid] = x[tid * NPIX + pix];

    float regSum = 0.0f;   // sum of log(w) + log(1-w) over this lane's (b,p) set

    #pragma unroll
    for (int layer = 0; layer < 4; ++layer) {
        // sigmoid of the 16 toggle gates for this layer & pixel
        if (tid < NPAT) {
            float g = gates[(layer * NPAT + tid) * NPIX + pix];
            tvals[tid] = (_Float16)(1.0f / (1.0f + __expf(-g)));
        }
        if (layer < 3) {
            // warm GL2 for next layer's gates (global_prefetch_b8 path)
            __builtin_prefetch(&gates[((layer + 1) * NPAT) * NPIX + pix], 0, 1);
        }
        __syncthreads();

        const float* cur = buf[layer & 1];
        float*       nxt = buf[(layer + 1) & 1];

        // B operand (32x16 f16, K=16..31 zero-pad): elements 0..7 carry gate
        // values K=0..7 (lanes 0-15) / K=8..15 (lanes 16-31); elements 8..15 zero.
        v16h bmat = {};
        {
            #pragma unroll
            for (int k = 0; k < 8; ++k) bmat[k] = tvals[halfSel * 8 + k];
        }

        v8f acc0 = {};   // rows waveBase + 0..15
        v8f acc1 = {};   // rows waveBase + 16..31

        #pragma unroll
        for (int half = 0; half < 2; ++half) {
            const int bb = waveBase + half * 16 + rowInTile;   // b-row this lane feeds
            const float v0 = cur[bb];
            const float v1 = cur[(bb + 1) & (NB - 1)];         // shifts 1 and 1 (v2==v1)
            const float v3 = cur[(bb + 2) & (NB - 1)];

            // pattern factor for bit0 (j=0): lanes>=16 handle p=8..15 (bit0=1)
            const float f0 = halfSel ? v0 : (1.0f - v0);

            // combos over j=1 (v1) and j=2 (v2==v1)
            float g12[4];
            g12[0] = (1.0f - v1) * (1.0f - v1);   // b1=0,b2=0
            g12[1] = (1.0f - v1) * v1;            // b1=0,b2=1
            g12[2] = v1 * (1.0f - v1);            // b1=1,b2=0
            g12[3] = v1 * v1;                     // b1=1,b2=1
            float a3[2] = {1.0f - v3, v3};

            // A operand (16x32 f16): this lane's row M=rowInTile, its 8 K's in
            // elements 0..7 (K=0..7 for lanes 0-15, K=8..15 for lanes 16-31),
            // elements 8..15 (K=16..23 / 24..31) are zero padding.
            v16h amat = {};
            #pragma unroll
            for (int q = 0; q < 8; ++q) {
                float w = f0 * g12[q >> 1] * a3[q & 1];
                amat[q] = (_Float16)w;
                if (layer > 0) {
                    regSum += __logf(w) + __logf(1.0f - w);
                }
            }

            if (half == 0) {
                acc0 = __builtin_amdgcn_wmma_f32_16x16x32_f16(
                    false, amat, false, bmat, (short)0, acc0, false, false);
            } else {
                acc1 = __builtin_amdgcn_wmma_f32_16x16x32_f16(
                    false, amat, false, bmat, (short)0, acc1, false, false);
            }
        }

        // All D columns identical (B broadcast). D layout: lanes 0-15 hold
        // M=r in VGPR r; lanes 16-31 hold M=r+8. Lanes 0 and 16 scatter.
        if (lane == 0) {
            #pragma unroll
            for (int r = 0; r < 8; ++r) {
                nxt[waveBase + r]      = clamp01(acc0[r]);
                nxt[waveBase + 16 + r] = clamp01(acc1[r]);
            }
        } else if (lane == 16) {
            #pragma unroll
            for (int r = 0; r < 8; ++r) {
                nxt[waveBase + 8 + r]  = clamp01(acc0[r]);
                nxt[waveBase + 24 + r] = clamp01(acc1[r]);
            }
        }
        __syncthreads();
    }

    // Final outputs are in buf[0] (4 layers -> buffer index 4&1 == 0)
    out[tid * NPIX + pix] = buf[0][tid];

    // Deterministic block reduction of regSum
    red[tid] = regSum;
    __syncthreads();
    #pragma unroll
    for (int s = 128; s > 0; s >>= 1) {
        if (tid < s) red[tid] += red[tid + s];
        __syncthreads();
    }
    if (tid == 0) blocksum[pix] = red[0];
}

// Single-block deterministic final reduction:
// reg/3 = -(sum over layers 1..3, b, hw, p of log(w)+log(1-w)) / (3 * 2^26)
__global__ __launch_bounds__(256) void lgn_reduce_kernel(
    const float* __restrict__ blocksum, float* __restrict__ out_reg)
{
    __shared__ float red[256];
    const int tid = threadIdx.x;
    float s = 0.0f;
    for (int i = tid; i < NPIX; i += 256) s += blocksum[i];
    red[tid] = s;
    __syncthreads();
    #pragma unroll
    for (int st = 128; st > 0; st >>= 1) {
        if (tid < st) red[tid] += red[tid + st];
        __syncthreads();
    }
    if (tid == 0) {
        out_reg[0] = -red[0] / (3.0f * 67108864.0f);  // 2^26 elements per mean
    }
}

extern "C" void kernel_launch(void* const* d_in, const int* in_sizes, int n_in,
                              void* d_out, int out_size, void* d_ws, size_t ws_size,
                              hipStream_t stream) {
    (void)in_sizes; (void)n_in; (void)out_size; (void)ws_size;
    const float* x     = (const float*)d_in[0];   // (256,64,64)
    const float* gates = (const float*)d_in[1];   // (4,16,64,64)
    float* out     = (float*)d_out;               // 1048576 outputs + 1 reg scalar
    float* out_reg = out + (NB * NPIX);
    float* blocksum = (float*)d_ws;               // 4096 floats of scratch

    lgn_forward_kernel<<<NPIX, NB, 0, stream>>>(x, gates, out, blocksum);
    lgn_reduce_kernel<<<1, 256, 0, stream>>>(blocksum, out_reg);
}
